// AttentionLayer_28278064677580
// MI455X (gfx1250) — matrix operand order repaired and checked by reading, not compiled, not run on hardware
//
#include <hip/hip_runtime.h>
#include <hip/hip_bf16.h>

// ---------------------------------------------------------------------------
// Linear-attention layer on MI455X (gfx1250), wave32 + v_wmma_f32_16x16x32_bf16
// + Tensor Data Mover staging + global_prefetch.
//
// Roofline: ~34 GFLOP total, ~100MB HBM traffic (4.3us @ 23.3TB/s) -> all three
// GEMMs (conv-as-GEMM, k^T v, q*(kv)) go through the WMMA pipes, bf16 operands
// with f32 accumulate; phi/bias/GELU/residual in fp32.
//
// WMMA 16x16x32 bf16 fragment layouts (ISA 7.12.2, wave32):
//   A (16x32): lane l -> row m = l&15; K elems {h*8+j} and {16+h*8+j}, h=l>>4
//              => two contiguous 16B loads from row-major [M][K] memory.
//   B (32x16): lane l -> col n = l&15; K elems {h*8+j} and {16+h*8+j}, h=l>>4
//              => two contiguous 16B loads from column-major ([N][K]) memory.
//   C/D f32  : lane l -> col n = l&15; VGPR r -> row m = (l>>4)*8 + r.
// ---------------------------------------------------------------------------

#define DEV __device__ __forceinline__

typedef __attribute__((ext_vector_type(16))) __bf16 v16bf;
typedef __attribute__((ext_vector_type(8)))  __bf16 v8bf;
typedef __attribute__((ext_vector_type(8)))  float  v8f;
typedef __attribute__((ext_vector_type(4)))  unsigned int v4u;
typedef __attribute__((ext_vector_type(4)))  int v4i;
typedef __attribute__((ext_vector_type(8)))  int v8i;

constexpr int Bb = 8;     // batch
constexpr int Cc = 256;   // channels
constexpr int Nn = 4096;  // sequence
constexpr int CO = 512;   // conv out channels (2C)
constexpr int NP = Nn + 2;  // padded sequence (zero row at n=-1 and n=N)

DEV v8f wmma_bf16(v16bf a, v16bf b, v8f c) {
  return __builtin_amdgcn_wmma_f32_16x16x32_bf16(
      /*neg_a=*/false, a, /*neg_b=*/false, b,
      /*c_mod=*/(short)0, c, /*reuse_a=*/false, /*reuse_b=*/false);
}

// A fragment from row-major bf16 [M][K]; rowk0 points at A[m][k0]
DEV v16bf frag_a_bf16(const __bf16* rowk0, int half) {
  union { v16bf v; v8bf h[2]; } u;
  u.h[0] = *reinterpret_cast<const v8bf*>(rowk0 + half * 8);
  u.h[1] = *reinterpret_cast<const v8bf*>(rowk0 + 16 + half * 8);
  return u.v;
}
// A fragment from row-major f32 [M][K] (LDS or global) with bf16 convert
DEV v16bf frag_a_f32(const float* rowk0, int half) {
  const float* p = (const float*)__builtin_assume_aligned(rowk0 + half * 8, 32);
  const float* q = (const float*)__builtin_assume_aligned(rowk0 + 16 + half * 8, 32);
  v16bf v;
#pragma unroll
  for (int j = 0; j < 8; ++j) { v[j] = (__bf16)p[j]; v[j + 8] = (__bf16)q[j]; }
  return v;
}
// B fragment from column-major bf16 ([N][K] memory, K contiguous)
DEV v16bf frag_b_bf16(const __bf16* colk0, int half) {
  union { v16bf v; v8bf h[2]; } u;
  u.h[0] = *reinterpret_cast<const v8bf*>(colk0 + half * 8);
  u.h[1] = *reinterpret_cast<const v8bf*>(colk0 + 16 + half * 8);
  return u.v;
}

// ---------------------------------------------------------------------------
// Kernel 1: repack conv weights [512][256][3] f32 -> [3][512][256] bf16
// ---------------------------------------------------------------------------
__global__ __launch_bounds__(256) void k_pack_w(const float* __restrict__ w,
                                                __bf16* __restrict__ wbf) {
  int idx = blockIdx.x * 256 + threadIdx.x;
  if (idx >= CO * Cc * 3) return;
  int t = idx % 3;
  int c = (idx / 3) % Cc;
  int o = idx / (3 * Cc);
  wbf[((size_t)t * CO + o) * Cc + c] = (__bf16)w[idx];
}

// ---------------------------------------------------------------------------
// Kernel 2a: zero the two padding rows of xTp (rows 0 and N+1 per batch)
// ---------------------------------------------------------------------------
__global__ __launch_bounds__(256) void k_zero_pad(__bf16* __restrict__ xTp) {
  int idx = blockIdx.x * 256 + threadIdx.x;  // Bb*2*Cc = 4096
  if (idx >= Bb * 2 * Cc) return;
  int c = idx & 255;
  int which = (idx >> 8) & 1;
  int b = idx >> 9;
  size_t row = (size_t)b * NP + (which ? (Nn + 1) : 0);
  xTp[row * Cc + c] = (__bf16)0.f;
}

// ---------------------------------------------------------------------------
// Kernel 2b: x [B][C][N] f32 -> xTp [B][N+2][C] bf16 (interior rows, +1 shift)
// via LDS 32x32 tile transpose
// ---------------------------------------------------------------------------
__global__ __launch_bounds__(256) void k_xT(const float* __restrict__ x,
                                            __bf16* __restrict__ xTp) {
  __shared__ __bf16 tile[32 * 33];
  int bid = blockIdx.x;
  int ntl = bid & 127;          // 4096/32
  int ctl = (bid >> 7) & 7;     // 256/32
  int b   = bid >> 10;
  int n0 = ntl * 32, c0 = ctl * 32;
  int r0 = threadIdx.x >> 5, lc = threadIdx.x & 31;
#pragma unroll
  for (int i = 0; i < 4; ++i) {
    int cl = r0 + i * 8;
    float v = x[((size_t)b * Cc + c0 + cl) * Nn + n0 + lc];
    tile[cl * 33 + lc] = (__bf16)v;
  }
  __syncthreads();
#pragma unroll
  for (int i = 0; i < 4; ++i) {
    int nl = r0 + i * 8;
    xTp[((size_t)b * NP + n0 + nl + 1) * Cc + c0 + lc] = tile[lc * 33 + nl];
  }
}

// ---------------------------------------------------------------------------
// Kernel 3: conv(k=3,pad=1) + bias + phi(elu+1).  Branchless via padded xTp.
// Wave tile: 64 output channels x 16 positions; per K-block: 3 tap B-fragments
// loaded up front, then 12 WMMAs (B reused 4x).  96 WMMAs per wave.
// q-half (o<256)  -> qph [B][N][C] bf16 (16B packed store per lane)
// k-half (o>=256) -> kph [B][C][N] bf16 via per-wave LDS transpose
// ---------------------------------------------------------------------------
__global__ __launch_bounds__(256) void k_conv_phi(const __bf16* __restrict__ xTp,
                                                  const __bf16* __restrict__ wbf,
                                                  const float* __restrict__ bias,
                                                  __bf16* __restrict__ qph,
                                                  __bf16* __restrict__ kph) {
  __shared__ float ts[8][16 * 17];  // per-wave 16x16 transpose scratch (+pad)
  const int wid = threadIdx.x >> 5, lane = threadIdx.x & 31;
  const int w = blockIdx.x * 8 + wid;          // 16384 wave tasks
  const int nt = w & 255, ot = (w >> 8) & 7, b = w >> 11;
  const int n0 = nt * 16, o_base = ot * 64;
  const int col = lane & 15, half = lane >> 4;

  v8f acc[4];
#pragma unroll
  for (int os = 0; os < 4; ++os)
#pragma unroll
    for (int r = 0; r < 8; ++r) acc[os][r] = 0.f;

  // padded row (n0+col+t) corresponds to original n = n0+col+t-1
  const __bf16* bbase = xTp + ((size_t)b * NP + n0 + col) * Cc;

  for (int cb = 0; cb < 8; ++cb) {             // K = 256 in blocks of 32
    v16bf bfr[3];
#pragma unroll
    for (int t = 0; t < 3; ++t)
      bfr[t] = frag_b_bf16(bbase + (size_t)t * Cc + cb * 32, half);
#pragma unroll
    for (int t = 0; t < 3; ++t) {
#pragma unroll
      for (int os = 0; os < 4; ++os) {
        const __bf16* ap =
            wbf + ((size_t)t * CO + o_base + os * 16 + col) * Cc + cb * 32;
        acc[os] = wmma_bf16(frag_a_bf16(ap, half), bfr[t], acc[os]);
      }
    }
  }

  const bool qhalf = (o_base < 256);           // uniform per block
#pragma unroll 1
  for (int os = 0; os < 4; ++os) {
    const int o_t = o_base + os * 16;
    const float* bp =
        (const float*)__builtin_assume_aligned(bias + o_t + half * 8, 32);
    float vals[8];
#pragma unroll
    for (int r = 0; r < 8; ++r) {
      float y = acc[os][r] + bp[r];
      vals[r] = (y > 0.f) ? (y + 1.f) : expf(y);   // elu(y)+1
    }
    if (qhalf) {
      v8bf pk;
#pragma unroll
      for (int r = 0; r < 8; ++r) pk[r] = (__bf16)vals[r];
      *reinterpret_cast<v8bf*>(qph + ((size_t)b * Nn + n0 + col) * Cc + o_t +
                               half * 8) = pk;
    } else {
      float* t = ts[wid];
#pragma unroll
      for (int r = 0; r < 8; ++r) t[(half * 8 + r) * 17 + col] = vals[r];
      // per-wave LDS is in-order; wait + clobber keeps compiler honest
      asm volatile("s_wait_dscnt 0" ::: "memory");
      v8bf pk;
#pragma unroll
      for (int j = 0; j < 8; ++j) pk[j] = (__bf16)t[col * 17 + half * 8 + j];
      asm volatile("" ::: "memory");
      *reinterpret_cast<v8bf*>(kph + ((size_t)b * Cc + (o_t - 256) + col) * Nn +
                               n0 + half * 8) = pk;
    }
  }
}

// ---------------------------------------------------------------------------
// Kernel 4: kvT[b][d][c] = sum_n x[b][d][n] * kph[b][c][n]   (K = 4096)
// dt and b are block-uniform, so the 16-row A-slab of x is shared by all 8
// waves: wave 0 stages it in LDS with the Tensor Data Mover (one 16x256 f32
// 2-D tile per 256-wide K-chunk, D# per ISA 8.3/8.4), s_wait_tensorcnt, then
// barrier.  B columns stream from global with a one-chunk-ahead prefetch
// (global_prefetch_b8).  8 WMMAs per chunk per wave, 128 total per tile.
// Stored transposed as bf16 => contiguous B operand of k_out.
// ---------------------------------------------------------------------------
__global__ __launch_bounds__(256) void k_kv(const float* __restrict__ x,
                                            const __bf16* __restrict__ kph,
                                            __bf16* __restrict__ kvT) {
  __shared__ __align__(16) float axt[16 * 256];   // A tile: 16 rows x 256 n
  const int wid = threadIdx.x >> 5, lane = threadIdx.x & 31;
  const int w = blockIdx.x * 8 + wid;          // 2048 wave tasks
  const int ct = w & 15, dt = (w >> 4) & 15, b = w >> 8;  // dt,b uniform/block
  const int c0 = ct * 16, d0 = dt * 16;
  const int col = lane & 15, half = lane >> 4;

  v8f acc;
#pragma unroll
  for (int r = 0; r < 8; ++r) acc[r] = 0.f;

  const __bf16* bcol = kph + ((size_t)b * Cc + c0 + col) * Nn;  // B col n=c
  const float* aslab = x + ((size_t)b * Cc + d0) * Nn;          // 16 d-rows

  for (int kc = 0; kc < 16; ++kc) {            // K chunks of 256
    if (wid == 0) {
      // ---- Tensor DMA descriptor (D#) -----------------------------------
      // group0: count=1 | lds_addr | global_addr[56:0] | type=2
      unsigned long long ga = (unsigned long long)(size_t)(aslab + kc * 256);
      unsigned int ldsa = (unsigned int)(size_t)(&axt[0]);  // LDS byte offset
      v4u g0 = { 1u, ldsa, (unsigned int)ga,
                 (unsigned int)(ga >> 32) | (2u << 30) };
      // group1: data_size=4B(code 2); tensor_dim0=4096; tensor_dim1=16;
      //         tile_dim0=256; tile_dim1=16; tensor_dim0_stride=4096
      v8i g1 = { (int)(2u << 16),        // [17:16] data_size
                 (int)(4096u << 16),     // tensor_dim0[15:0] << 16
                 (int)(16u << 16),       // tensor_dim0 hi=0 | tensor_dim1 lo
                 (int)(256u << 16),      // tensor_dim1 hi=0 | tile_dim0
                 (int)16,                // tile_dim1=16, tile_dim2=0
                 (int)4096,              // tensor_dim0_stride lo32
                 0, 0 };
      v4i z4 = {0, 0, 0, 0};
      v8i z8 = {0, 0, 0, 0, 0, 0, 0, 0};
      // 6-arg variant (clang-23 / therock headers): groups 2,3,(aux), cpol
      __builtin_amdgcn_tensor_load_to_lds(g0, g1, z4, z4, z8, 0);
      __builtin_amdgcn_s_wait_tensorcnt(0);
    }
    __syncthreads();                     // publish LDS tile to all waves
    __builtin_prefetch(bcol + (kc + 1) * 256, 0, 0);  // next B chunk -> L2
#pragma unroll
    for (int kb = 0; kb < 8; ++kb) {
      const float* ar = &axt[col * 256 + kb * 32];    // A row m = d-local
      acc = wmma_bf16(frag_a_f32(ar, half),
                      frag_b_bf16(bcol + kc * 256 + kb * 32, half), acc);
    }
    __syncthreads();                     // tile consumed before next DMA
  }

#pragma unroll
  for (int r = 0; r < 8; ++r) {
    kvT[((size_t)b * Cc + d0 + half * 8 + r) * Cc + c0 + col] = (__bf16)acc[r];
  }
}

// ---------------------------------------------------------------------------
// Kernel 5: out[b][d][n] = gelu_exact( sum_c qph[b][n][c]*kvT[b][d][c] ) + x
// Lane owns 8 consecutive n for one channel d -> 32B residual load + 32B store.
// ---------------------------------------------------------------------------
__global__ __launch_bounds__(256) void k_out(const __bf16* __restrict__ qph,
                                             const __bf16* __restrict__ kvT,
                                             const float* __restrict__ x,
                                             float* __restrict__ out) {
  const int wid = threadIdx.x >> 5, lane = threadIdx.x & 31;
  const int w = blockIdx.x * 8 + wid;          // 32768 wave tasks
  const int dt = w & 15, nt = (w >> 4) & 255, b = w >> 12;
  const int d0 = dt * 16, n0 = nt * 16;
  const int col = lane & 15, half = lane >> 4;

  v8f acc;
#pragma unroll
  for (int r = 0; r < 8; ++r) acc[r] = 0.f;

  const __bf16* arow = qph + ((size_t)b * Nn + n0 + col) * Cc;  // A row m=n
  const __bf16* bcol = kvT + ((size_t)b * Cc + d0 + col) * Cc;  // B col n=d
#pragma unroll
  for (int cb = 0; cb < 8; ++cb) {
    acc = wmma_bf16(frag_a_bf16(arow + cb * 32, half),
                    frag_b_bf16(bcol + cb * 32, half), acc);
  }

  const int d = d0 + col;
  const size_t off = ((size_t)b * Cc + d) * Nn + n0 + half * 8;
  const float* xr = (const float*)__builtin_assume_aligned(x + off, 32);
  float* orow = (float*)__builtin_assume_aligned(out + off, 32);
#pragma unroll
  for (int r = 0; r < 8; ++r) {
    float v = acc[r];
    float g = 0.5f * v * (1.f + erff(v * 0.70710678118654752f));  // exact gelu
    orow[r] = g + xr[r];
  }
}

// ---------------------------------------------------------------------------
extern "C" void kernel_launch(void* const* d_in, const int* in_sizes, int n_in,
                              void* d_out, int out_size, void* d_ws,
                              size_t ws_size, hipStream_t stream) {
  const float* x  = (const float*)d_in[0];  // [8,256,4096]
  const float* cw = (const float*)d_in[1];  // [512,256,3]
  const float* cb = (const float*)d_in[2];  // [512]
  float* out = (float*)d_out;               // [8,256,4096]

  char* ws = (char*)d_ws;
  const size_t SZ_XTP = (size_t)Bb * NP * Cc * 2;   // padded transpose, bf16
  const size_t SZ16   = (size_t)Bb * Nn * Cc * 2;   // 16 MiB bf16 tensor
  __bf16* xTp = (__bf16*)(ws);                          // [B][N+2][C]
  __bf16* qph = (__bf16*)(ws + SZ_XTP);                 // [B][N][C]
  __bf16* kph = (__bf16*)(ws + SZ_XTP + SZ16);          // [B][C][N]
  __bf16* wbf = (__bf16*)(ws + SZ_XTP + 2 * SZ16);      // [3][512][256]
  __bf16* kvT = (__bf16*)(ws + SZ_XTP + 2 * SZ16 +
                          (size_t)3 * CO * Cc * 2);     // [B][C][C]

  k_pack_w<<<(3 * CO * Cc + 255) / 256, 256, 0, stream>>>(cw, wbf);
  k_zero_pad<<<(Bb * 2 * Cc + 255) / 256, 256, 0, stream>>>(xTp);
  k_xT<<<Bb * (Cc / 32) * (Nn / 32), 256, 0, stream>>>(x, xTp);
  k_conv_phi<<<(Bb * 8 * (Nn / 16)) / 8, 256, 0, stream>>>(xTp, wbf, cb, qph, kph);
  k_kv<<<(Bb * 16 * 16) / 8, 256, 0, stream>>>(x, kph, kvT);
  k_out<<<(Bb * 16 * (Nn / 16)) / 8, 256, 0, stream>>>(qph, kvT, x, out);
}
